// AttentionSoftmaxPlusOne_57913339019487
// MI455X (gfx1250) — matrix operand order repaired and checked by reading, not compiled, not run on hardware
//
#include <hip/hip_runtime.h>

#define DIM    1024
#define NHEADS 16
#define HD     64
#define SEQ    2048
#define BATCH  2
#define BR     64
#define BC     32

#if __has_builtin(__builtin_amdgcn_global_load_async_to_lds_b128) && \
    __has_builtin(__builtin_amdgcn_s_wait_asynccnt)
#define USE_ASYNC_LDS 1
#endif

typedef __attribute__((ext_vector_type(16))) __bf16 v16bf;
typedef __attribute__((ext_vector_type(8)))  float  v8f;
typedef __attribute__((ext_vector_type(4)))  int    v4i;

typedef __attribute__((address_space(1))) v4i* as1_v4i_ptr;
typedef __attribute__((address_space(3))) v4i* as3_v4i_ptr;

union BFrag {
    v16bf v;
    uint4 q[2];
    unsigned short h[16];
};

__device__ __forceinline__ unsigned short f32_to_bf16(float f) {
    unsigned int u = __float_as_uint(f);
    u += 0x7FFFu + ((u >> 16) & 1u);   // round-to-nearest-even
    return (unsigned short)(u >> 16);
}

__device__ __forceinline__ v8f v8f_zero() {
    v8f z;
#pragma unroll
    for (int i = 0; i < 8; ++i) z[i] = 0.0f;
    return z;
}

// ---------------------------------------------------------------------------
// f32 -> bf16 conversion (4 elements/thread)
// ---------------------------------------------------------------------------
__global__ void cvt_bf16_kernel(const float* __restrict__ in,
                                unsigned short* __restrict__ out, int n) {
    int i = (blockIdx.x * blockDim.x + threadIdx.x) * 4;
    if (i < n) {
        float4 f = *(const float4*)(in + i);
        ushort4 o;
        o.x = f32_to_bf16(f.x);
        o.y = f32_to_bf16(f.y);
        o.z = f32_to_bf16(f.z);
        o.w = f32_to_bf16(f.w);
        *(ushort4*)(out + i) = o;
    }
}

// ---------------------------------------------------------------------------
// C[M,N] = alpha * A[M,K] @ B[N,K]^T (+ bias); A,B bf16 row-major.
// Output: bf16 (Cb) or f32 (Cf, with bias).
// Block: 128 threads = 4 waves; wave tile 32x64; block tile 64x128.
// ---------------------------------------------------------------------------
__global__ __launch_bounds__(128)
void gemm_bf16_nt(const unsigned short* __restrict__ A,
                  const unsigned short* __restrict__ B,
                  const float* __restrict__ bias,
                  unsigned short* __restrict__ Cb,
                  float* __restrict__ Cf,
                  int M, int N, int K, float alpha)
{
    const int lane = threadIdx.x & 31;
    const int wid  = threadIdx.x >> 5;
    const int l16  = lane & 15;
    const int hsel = lane >> 4;        // 0 or 1 (half-wave select)
    const int wm   = wid >> 1, wn = wid & 1;
    const int m0   = blockIdx.y * 64  + wm * 32;
    const int n0   = blockIdx.x * 128 + wn * 64;

    v8f acc[2][4];
#pragma unroll
    for (int mi = 0; mi < 2; ++mi)
#pragma unroll
        for (int ni = 0; ni < 4; ++ni) acc[mi][ni] = v8f_zero();

    for (int k0 = 0; k0 < K; k0 += 32) {
        BFrag a[2], b[4];
#pragma unroll
        for (int mi = 0; mi < 2; ++mi) {
            // A-frag: lane row = m0+mi*16+l16; elems 0..7 at K=k0+hsel*8, 8..15 at +16
            const unsigned short* ap =
                A + (size_t)(m0 + mi * 16 + l16) * K + k0 + hsel * 8;
            a[mi].q[0] = *(const uint4*)ap;
            a[mi].q[1] = *(const uint4*)(ap + 16);
        }
#pragma unroll
        for (int ni = 0; ni < 4; ++ni) {
            // B-frag: lane col = n0+ni*16+l16; elems 0..7 at K=k0+hsel*8, 8..15 at +16
            const unsigned short* bp =
                B + (size_t)(n0 + ni * 16 + l16) * K + k0 + hsel * 8;
            b[ni].q[0] = *(const uint4*)bp;
            b[ni].q[1] = *(const uint4*)(bp + 16);
        }
#pragma unroll
        for (int mi = 0; mi < 2; ++mi)
#pragma unroll
            for (int ni = 0; ni < 4; ++ni)
                acc[mi][ni] = __builtin_amdgcn_wmma_f32_16x16x32_bf16(
                    false, a[mi].v, false, b[ni].v, (short)0, acc[mi][ni],
                    false, false);
    }

#pragma unroll
    for (int mi = 0; mi < 2; ++mi)
#pragma unroll
        for (int ni = 0; ni < 4; ++ni)
#pragma unroll
            for (int r = 0; r < 8; ++r) {
                int row = m0 + mi * 16 + hsel * 8 + r;
                int col = n0 + ni * 16 + l16;
                float vv = acc[mi][ni][r] * alpha;
                if (Cf)
                    Cf[(size_t)row * N + col] = vv + (bias ? bias[col] : 0.0f);
                else
                    Cb[(size_t)row * N + col] = f32_to_bf16(vv);
            }
}

// ---------------------------------------------------------------------------
// Causal flash attention with softmax+1 denominator.
// Grid: (SEQ/BR, NHEADS, BATCH); block 128 (4 waves, 16 Q-rows each).
// Q/K/V/O are bf16 [BATCH*SEQ, DIM]; head h occupies columns h*HD..h*HD+63.
// Q is pre-scaled by 1/sqrt(HD).
// ---------------------------------------------------------------------------
__global__ __launch_bounds__(128)
void flash_attn_kernel(const unsigned short* __restrict__ Q,
                       const unsigned short* __restrict__ Km,
                       const unsigned short* __restrict__ V,
                       const float* __restrict__ denom_bias,
                       unsigned short* __restrict__ O)
{
    __shared__ __align__(16) unsigned short Kt[BC * HD];      // [j][d] row-major
    __shared__ __align__(16) unsigned short Vt[HD * BC];      // [d][j] transposed
    __shared__ __align__(16) unsigned short Pb[4][16 * BC];   // per-wave P staging

    const int i0   = blockIdx.x * BR;
    const int h    = blockIdx.y;
    const int bidx = blockIdx.z;
    const size_t hbase = (size_t)bidx * SEQ * DIM + (size_t)h * HD;
    const unsigned short* Qh = Q  + hbase;
    const unsigned short* Kh = Km + hbase;
    const unsigned short* Vh = V  + hbase;
    unsigned short*       Oh = O  + hbase;

    const int tid  = threadIdx.x;
    const int lane = tid & 31;
    const int wid  = tid >> 5;
    const int l16  = lane & 15;
    const int hsel = lane >> 4;

    // Resident Q fragments: rows i0 + wid*16 + l16, K (=head dim) 0..63
    BFrag qf[2];
    {
        const unsigned short* qrow = Qh + (size_t)(i0 + wid * 16 + l16) * DIM;
#pragma unroll
        for (int kk = 0; kk < 2; ++kk) {
            int kb = kk * 32 + hsel * 8;
            qf[kk].q[0] = *(const uint4*)(qrow + kb);
            qf[kk].q[1] = *(const uint4*)(qrow + kb + 16);
        }
    }

    float m[8], l[8];
    v8f acc[4];
#pragma unroll
    for (int r = 0; r < 8; ++r) { m[r] = -1e30f; l[r] = 0.0f; }
#pragma unroll
    for (int nt = 0; nt < 4; ++nt) acc[nt] = v8f_zero();

    const int row_base = i0 + wid * 16 + hsel * 8;   // row of acc element r = row_base + r

    for (int j0 = 0; j0 < i0 + BR; j0 += BC) {
        __syncthreads();
        // ---- stage K tile (row-major) into LDS ----
#ifdef USE_ASYNC_LDS
        // gfx1250 async DMA path: LDS written directly, tracked by ASYNCcnt
        for (int t = tid; t < (BC * HD) / 8; t += 128) {
            int jr = t >> 3;
            int dc = (t & 7) * 8;
            __builtin_amdgcn_global_load_async_to_lds_b128(
                (as1_v4i_ptr)(Kh + (size_t)(j0 + jr) * DIM + dc),
                (as3_v4i_ptr)&Kt[jr * HD + dc],
                0, 0);
        }
#else
        for (int t = tid; t < (BC * HD) / 8; t += 128) {
            int jr = t >> 3;
            int dc = (t & 7) * 8;
            *(uint4*)&Kt[jr * HD + dc] =
                *(const uint4*)(Kh + (size_t)(j0 + jr) * DIM + dc);
        }
#endif
        // ---- stage V tile transposed into LDS (needs element shuffle) ----
        for (int t = tid; t < (BC * HD) / 8; t += 128) {
            int jr = t >> 3;
            int dc = (t & 7) * 8;
            uint4 vv = *(const uint4*)(Vh + (size_t)(j0 + jr) * DIM + dc);
            const unsigned short* e = (const unsigned short*)&vv;
#pragma unroll
            for (int u = 0; u < 8; ++u) Vt[(dc + u) * BC + jr] = e[u];
        }
#ifdef USE_ASYNC_LDS
        __builtin_amdgcn_s_wait_asynccnt(0);
#endif
        __syncthreads();

        // ---- prefetch next tile's K/V rows into cache while we compute ----
        {
            int jn = j0 + BC;
            if (jn < i0 + BR && tid < BC) {
                __builtin_prefetch(Kh + (size_t)(jn + tid) * DIM, 0, 3);
                __builtin_prefetch(Vh + (size_t)(jn + tid) * DIM, 0, 3);
            }
        }

        // ---- S (16x32) = Q @ K^T ----
        v8f s[2];
#pragma unroll
        for (int nt = 0; nt < 2; ++nt) {
            v8f sa = v8f_zero();
#pragma unroll
            for (int kk = 0; kk < 2; ++kk) {
                BFrag kf;
                const unsigned short* kp =
                    &Kt[(nt * 16 + l16) * HD + kk * 32 + hsel * 8];
                kf.q[0] = *(const uint4*)kp;
                kf.q[1] = *(const uint4*)(kp + 16);
                sa = __builtin_amdgcn_wmma_f32_16x16x32_bf16(
                    false, qf[kk].v, false, kf.v, (short)0, sa, false, false);
            }
            s[nt] = sa;
        }

        // ---- causal mask + online softmax (per row r, across 16-lane half) ----
#pragma unroll
        for (int r = 0; r < 8; ++r) {
            int rowg = row_base + r;
            float s0 = s[0][r]; if (j0 + l16      > rowg) s0 = -1e30f;
            float s1 = s[1][r]; if (j0 + 16 + l16 > rowg) s1 = -1e30f;
            float t = fmaxf(s0, s1);
#pragma unroll
            for (int off = 8; off; off >>= 1)
                t = fmaxf(t, __shfl_xor(t, off, 32));
            float mn = fmaxf(m[r], t);
            float sc = __expf(m[r] - mn);
            float p0 = __expf(s0 - mn);
            float p1 = __expf(s1 - mn);
            float ls = p0 + p1;
#pragma unroll
            for (int off = 8; off; off >>= 1)
                ls += __shfl_xor(ls, off, 32);
            l[r] = l[r] * sc + ls;
            m[r] = mn;
#pragma unroll
            for (int nt = 0; nt < 4; ++nt) acc[nt][r] *= sc;
            // stage P (bf16) into per-wave LDS in C-layout positions
            int rl = hsel * 8 + r;
            Pb[wid][rl * BC + l16]      = f32_to_bf16(p0);
            Pb[wid][rl * BC + 16 + l16] = f32_to_bf16(p1);
        }

        // ---- read P back as A-fragment; acc += P @ V ----
        // (same-wave LDS write->read: DS ops are in-order within a wave)
        BFrag pf;
        {
            const unsigned short* pp = &Pb[wid][l16 * BC + hsel * 8];
            pf.q[0] = *(const uint4*)pp;
            pf.q[1] = *(const uint4*)(pp + 16);
        }
#pragma unroll
        for (int nt = 0; nt < 4; ++nt) {
            BFrag vf;
            const unsigned short* vp = &Vt[(nt * 16 + l16) * BC + hsel * 8];
            vf.q[0] = *(const uint4*)vp;
            vf.q[1] = *(const uint4*)(vp + 16);
            acc[nt] = __builtin_amdgcn_wmma_f32_16x16x32_bf16(
                false, pf.v, false, vf.v, (short)0, acc[nt], false, false);
        }
    }

    // ---- epilogue: divide by (l + exp(denom_bias[h])), store bf16 ----
    float eb = __expf(denom_bias[h]);
#pragma unroll
    for (int r = 0; r < 8; ++r) {
        float inv = 1.0f / (l[r] + eb);
        int rowg = row_base + r;
#pragma unroll
        for (int nt = 0; nt < 4; ++nt)
            Oh[(size_t)rowg * DIM + nt * 16 + l16] =
                f32_to_bf16(acc[nt][r] * inv);
    }
}

// ---------------------------------------------------------------------------
extern "C" void kernel_launch(void* const* d_in, const int* in_sizes, int n_in,
                              void* d_out, int out_size, void* d_ws, size_t ws_size,
                              hipStream_t stream)
{
    const float* x          = (const float*)d_in[0];
    const float* Wq         = (const float*)d_in[1];
    const float* Wk         = (const float*)d_in[2];
    const float* Wv         = (const float*)d_in[3];
    const float* Wo         = (const float*)d_in[4];
    const float* bo         = (const float*)d_in[5];
    const float* denom_bias = (const float*)d_in[6];

    const int    Mrows = BATCH * SEQ;              // 4096
    const size_t nX = (size_t)Mrows * DIM;         // 4M elements
    const size_t nW = (size_t)DIM * DIM;           // 1M elements

    // Workspace layout (bf16 buffers, ~48 MB total)
    unsigned short* ws  = (unsigned short*)d_ws;
    unsigned short* xb  = ws;
    unsigned short* wqb = xb  + nX;
    unsigned short* wkb = wqb + nW;
    unsigned short* wvb = wkb + nW;
    unsigned short* wob = wvb + nW;
    unsigned short* qb  = wob + nW;
    unsigned short* kb  = qb  + nX;
    unsigned short* vb  = kb  + nX;
    unsigned short* aob = vb  + nX;

    // 1) convert inputs to bf16
    cvt_bf16_kernel<<<dim3((unsigned)((nX / 4 + 255) / 256)), dim3(256), 0, stream>>>(x,  xb,  (int)nX);
    cvt_bf16_kernel<<<dim3((unsigned)((nW / 4 + 255) / 256)), dim3(256), 0, stream>>>(Wq, wqb, (int)nW);
    cvt_bf16_kernel<<<dim3((unsigned)((nW / 4 + 255) / 256)), dim3(256), 0, stream>>>(Wk, wkb, (int)nW);
    cvt_bf16_kernel<<<dim3((unsigned)((nW / 4 + 255) / 256)), dim3(256), 0, stream>>>(Wv, wvb, (int)nW);
    cvt_bf16_kernel<<<dim3((unsigned)((nW / 4 + 255) / 256)), dim3(256), 0, stream>>>(Wo, wob, (int)nW);

    // 2) projections: q = 0.125 * x@Wq^T (scale folded in), k, v
    dim3 gblk(128);
    dim3 ggrid(DIM / 128, Mrows / 64);             // (8, 64)
    gemm_bf16_nt<<<ggrid, gblk, 0, stream>>>(xb, wqb, nullptr, qb, nullptr,
                                             Mrows, DIM, DIM, 0.125f);
    gemm_bf16_nt<<<ggrid, gblk, 0, stream>>>(xb, wkb, nullptr, kb, nullptr,
                                             Mrows, DIM, DIM, 1.0f);
    gemm_bf16_nt<<<ggrid, gblk, 0, stream>>>(xb, wvb, nullptr, vb, nullptr,
                                             Mrows, DIM, DIM, 1.0f);

    // 3) causal flash attention with softmax+1
    dim3 fgrid(SEQ / BR, NHEADS, BATCH);           // (32, 16, 2)
    flash_attn_kernel<<<fgrid, dim3(128), 0, stream>>>(qb, kb, vb, denom_bias, aob);

    // 4) output projection: out = attn_out @ Wo^T + bo (f32)
    gemm_bf16_nt<<<ggrid, gblk, 0, stream>>>(aob, wob, bo, nullptr, (float*)d_out,
                                             Mrows, DIM, DIM, 1.0f);
}
